// MultiHeadSelfAttention_35777077576221
// MI455X (gfx1250) — 16-bit and 8-bit matrix operand order repaired and checked by reading, compiled, not run on hardware
//
#include <hip/hip_runtime.h>
#include <cmath>
#include <cstdint>
#include <cstddef>

typedef __attribute__((ext_vector_type(8)))  int      v8i;
typedef __attribute__((ext_vector_type(8)))  float    v8f;
typedef __attribute__((ext_vector_type(16))) _Float16 v16h;

#define HIDDEN 768
#define NHEAD  12
#define HDIM   64
#define SEQ    512
#define BATCH  8
#define QMAXF  127.0f

struct PLAParams {
    float m[12];
    float c[12];
    float itv[13];
};

__device__ __forceinline__ float wave_max(float v) {
    #pragma unroll
    for (int o = 16; o; o >>= 1) v = fmaxf(v, __shfl_xor(v, o, 32));
    return v;
}
__device__ __forceinline__ float wave_sum(float v) {
    #pragma unroll
    for (int o = 16; o; o >>= 1) v += __shfl_xor(v, o, 32);
    return v;
}
__device__ __forceinline__ signed char quant1(float v, float sc) {
    float p = rintf(v / sc);
    p = fminf(fmaxf(p, -QMAXF), QMAXF);
    return (signed char)(int)p;
}
__device__ __forceinline__ float fakequant(float v, float sc) {
    float p = rintf(v / sc);
    p = fminf(fmaxf(p, -QMAXF), QMAXF);
    return p * sc;
}
__device__ __forceinline__ int koffA8(int j, int h) {
    int jj = j & 3;
    return ((j >> 2) << 5) + ((jj >> 1) << 4) + ((jj & 1) << 2) + (h << 3);
}
__device__ __forceinline__ int koffB8(int j, int h) {
    return ((j >> 2) << 5) + (((j & 3) >> 1) << 4) + ((j & 1) << 2) + (h << 3);
}
__device__ __forceinline__ int koffA16(int j, int h) {
    return ((j >> 2) << 4) + ((j & 3) << 1) + (h << 3);
}

__global__ void rowquant768_kernel(const float* __restrict__ x,
                                   signed char* __restrict__ q,
                                   float* __restrict__ s) {
    const int r = blockIdx.x;
    const int t = threadIdx.x;
    const float* row = x + (long)r * HIDDEN;
    float v0 = row[t], v1 = row[t + 256], v2 = row[t + 512];
    float mx = fmaxf(fabsf(v0), fmaxf(fabsf(v1), fabsf(v2)));
    mx = wave_max(mx);
    __shared__ float red[8];
    const int lane = t & 31, w = t >> 5;
    if (lane == 0) red[w] = mx;
    __syncthreads();
    if (t == 0) {
        float m2 = red[0];
        #pragma unroll
        for (int i = 1; i < 8; ++i) m2 = fmaxf(m2, red[i]);
        float sc = m2 / QMAXF;
        if (sc == 0.0f) sc = 1.0f;
        red[0] = sc;
        s[r] = sc;
    }
    __syncthreads();
    const float sc = red[0];
    signed char* qr = q + (long)r * HIDDEN;
    qr[t]       = quant1(v0, sc);
    qr[t + 256] = quant1(v1, sc);
    qr[t + 512] = quant1(v2, sc);
}

__global__ void proj_iu8_kernel(const signed char* __restrict__ A,
                                const float* __restrict__ sA,
                                const signed char* __restrict__ W,
                                const float* __restrict__ sW,
                                const float* __restrict__ bias,
                                float* __restrict__ out) {
    const int lane = threadIdx.x & 31;
    const int wid  = threadIdx.x >> 5;
    const int lo   = lane & 15;
    const int hf   = lane >> 4;
    const int tm   = blockIdx.x * 16;
    const int tn0  = (blockIdx.y * 4 + wid) * 64;

    const int m = tm + lo;
    const signed char* arow = A + (long)m * HIDDEN;
    const signed char* wrow[4];
    #pragma unroll
    for (int u = 0; u < 4; ++u)
        wrow[u] = W + (long)(tn0 + u * 16 + lo) * HIDDEN;

    v8i acc0 = {}, acc1 = {}, acc2 = {}, acc3 = {};
    #pragma unroll
    for (int kc = 0; kc < HIDDEN / 64; ++kc) {
        const int kb = kc * 64;
        v8i af;
        #pragma unroll
        for (int j = 0; j < 8; ++j)
            af[j] = *(const int*)(arow + kb + koffA8(j, hf));
        v8i b0, b1, b2, b3;
        #pragma unroll
        for (int j = 0; j < 8; ++j) {
            const int ko = kb + koffB8(j, hf);
            b0[j] = *(const int*)(wrow[0] + ko);
            b1[j] = *(const int*)(wrow[1] + ko);
            b2[j] = *(const int*)(wrow[2] + ko);
            b3[j] = *(const int*)(wrow[3] + ko);
        }
        acc0 = __builtin_amdgcn_wmma_i32_16x16x64_iu8(true, af, true, b0, acc0, false, false);
        acc1 = __builtin_amdgcn_wmma_i32_16x16x64_iu8(true, af, true, b1, acc1, false, false);
        acc2 = __builtin_amdgcn_wmma_i32_16x16x64_iu8(true, af, true, b2, acc2, false, false);
        acc3 = __builtin_amdgcn_wmma_i32_16x16x64_iu8(true, af, true, b3, acc3, false, false);
    }

    float sam[8];
    #pragma unroll
    for (int r = 0; r < 8; ++r) sam[r] = sA[tm + r + hf * 8];

    v8i accs[4] = {acc0, acc1, acc2, acc3};
    #pragma unroll
    for (int u = 0; u < 4; ++u) {
        const int n = tn0 + u * 16 + lo;
        const float swn = sW[n];
        const float bn  = bias[n];
        #pragma unroll
        for (int r = 0; r < 8; ++r) {
            const int mm = tm + r + hf * 8;
            out[(long)mm * HIDDEN + n] = (float)accs[u][r] * sam[r] * swn + bn;
        }
    }
}

__global__ void headquant_i8_kernel(const float* __restrict__ xf,
                                    signed char* __restrict__ qi,
                                    float* __restrict__ sq) {
    const int lane = threadIdx.x & 31;
    const int wid  = threadIdx.x >> 5;
    const long g   = (long)blockIdx.x * 8 + wid;
    const int  h   = (int)(g % 12);
    const long bs  = g / 12;
    const int  b   = (int)(bs / SEQ);
    const int  s   = (int)(bs % SEQ);

    const float* src = xf + bs * HIDDEN + h * HDIM;
    const float a0 = src[lane], a1 = src[lane + 32];
    float mx = fmaxf(fabsf(a0), fabsf(a1));
    mx = wave_max(mx);
    float sc = mx / QMAXF;
    if (sc == 0.0f) sc = 1.0f;

    signed char* dst = qi + (((long)(b * NHEAD + h) * SEQ + s) * HDIM);
    dst[lane]      = quant1(a0, sc);
    dst[lane + 32] = quant1(a1, sc);
    if (lane == 0) sq[(long)(b * NHEAD + h) * SEQ + s] = sc;
}

__global__ void headquant_f16_kernel(const float* __restrict__ xf,
                                     _Float16* __restrict__ vh) {
    const int lane = threadIdx.x & 31;
    const int wid  = threadIdx.x >> 5;
    const long g   = (long)blockIdx.x * 8 + wid;
    const int  h   = (int)(g % 12);
    const long bs  = g / 12;
    const int  b   = (int)(bs / SEQ);
    const int  s   = (int)(bs % SEQ);

    const float* src = xf + bs * HIDDEN + h * HDIM;
    const float a0 = src[lane], a1 = src[lane + 32];
    float mx = fmaxf(fabsf(a0), fabsf(a1));
    mx = wave_max(mx);
    float sc = mx / QMAXF;
    if (sc == 0.0f) sc = 1.0f;

    _Float16* dst = vh + (((long)(b * NHEAD + h) * SEQ + s) * HDIM);
    dst[lane]      = (_Float16)fakequant(a0, sc);
    dst[lane + 32] = (_Float16)fakequant(a1, sc);
}

__global__ void scores_softmax_kernel(const signed char* __restrict__ qi,
                                      const float* __restrict__ sq,
                                      const signed char* __restrict__ ki,
                                      const float* __restrict__ sk,
                                      _Float16* __restrict__ ph,
                                      PLAParams pla) {
    __shared__ float srow[16][SEQ];

    const int lane = threadIdx.x;
    const int lo   = lane & 15;
    const int hf   = lane >> 4;
    const int qt   = blockIdx.x;
    const int bh   = blockIdx.y;

    const signed char* Q = qi + (long)bh * SEQ * HDIM;
    const signed char* K = ki + (long)bh * SEQ * HDIM;
    const float* SQ = sq + (long)bh * SEQ;
    const float* SK = sk + (long)bh * SEQ;

    const int m = qt * 16 + lo;
    v8i af;
    #pragma unroll
    for (int j = 0; j < 8; ++j)
        af[j] = *(const int*)(Q + (long)m * HDIM + koffA8(j, hf));

    float sqm[8];
    #pragma unroll
    for (int r = 0; r < 8; ++r) sqm[r] = SQ[qt * 16 + r + hf * 8];

    for (int nt = 0; nt < SEQ / 16; ++nt) {
        const int n = nt * 16 + lo;
        v8i bf;
        #pragma unroll
        for (int j = 0; j < 8; ++j)
            bf[j] = *(const int*)(K + (long)n * HDIM + koffB8(j, hf));
        v8i acc = {};
        acc = __builtin_amdgcn_wmma_i32_16x16x64_iu8(true, af, true, bf, acc,
                                                     false, false);
        const float skn = SK[n] * 0.125f;
        #pragma unroll
        for (int r = 0; r < 8; ++r)
            srow[r + hf * 8][n] = (float)acc[r] * sqm[r] * skn;
    }
    __syncthreads();

    for (int rr = 0; rr < 16; ++rr) {
        float mx = -3.402823e38f;
        #pragma unroll
        for (int t = 0; t < 16; ++t)
            mx = fmaxf(mx, srow[rr][lane + 32 * t]);
        const float rowmax = wave_max(mx);

        float evals[16];
        float sumE = 0.0f, maxE = 0.0f;
        #pragma unroll
        for (int t = 0; t < 16; ++t) {
            float sh = srow[rr][lane + 32 * t] - rowmax;
            float fx = rintf(sh * 67108864.0f);
            fx = fminf(fmaxf(fx, -2147483648.0f), 2147483647.0f);
            fx *= (1.0f / 67108864.0f);
            float xc = fminf(fmaxf(fx, -10.0f), 0.0f);
            int idx = -1;
            #pragma unroll
            for (int j = 0; j < 13; ++j) idx += (xc >= pla.itv[j]) ? 1 : 0;
            idx = min(max(idx, 0), 11);
            float e = pla.m[idx] * xc + pla.c[idx];
            evals[t] = e;
            sumE += e;
            maxE = fmaxf(maxE, e);
        }
        sumE = wave_sum(sumE);
        maxE = wave_max(maxE);

        const float inv = 1.0f / (sumE + 1e-9f);
        float sc = (maxE * inv) / QMAXF;
        if (sc == 0.0f) sc = 1.0f;

        _Float16* pr = ph + (long)bh * SEQ * SEQ + (long)(qt * 16 + rr) * SEQ;
        #pragma unroll
        for (int t = 0; t < 16; ++t)
            pr[lane + 32 * t] = (_Float16)fakequant(evals[t] * inv, sc);
    }
}

__global__ void ctx_f16_kernel(const _Float16* __restrict__ ph,
                               const _Float16* __restrict__ vh,
                               float* __restrict__ ctxf) {
    const int lane = threadIdx.x;
    const int lo   = lane & 15;
    const int hf   = lane >> 4;
    const int mt   = blockIdx.x;
    const int dt   = blockIdx.y;
    const int bh   = blockIdx.z;

    const _Float16* P = ph + (long)bh * SEQ * SEQ;
    const _Float16* V = vh + (long)bh * SEQ * HDIM;

    const int m = mt * 16 + lo;
    const int n = dt * 16 + lo;
    const _Float16* prow = P + (long)m * SEQ;

    v8f acc = {};
    for (int kc = 0; kc < SEQ / 32; ++kc) {
        const int kb = kc * 32;
        v16h a;
        #pragma unroll
        for (int j = 0; j < 8; ++j) {
            const int k0 = kb + koffA16(j, hf);
            a[2 * j]     = prow[k0];
            a[2 * j + 1] = prow[k0 + 1];
        }
        v16h b;
        #pragma unroll
        for (int j = 0; j < 8; ++j) {
            const int k0 = kb + koffA16(j, hf);
            b[2 * j]     = V[(long)k0 * HDIM + n];
            b[2 * j + 1] = V[(long)(k0 + 1) * HDIM + n];
        }
        acc = __builtin_amdgcn_wmma_f32_16x16x32_f16(false, a, false, b,
                                                     (short)0, acc, false, false);
    }

    const int bb = bh / NHEAD, hh = bh % NHEAD;
    float* dst = ctxf + (long)bb * SEQ * HIDDEN;
    #pragma unroll
    for (int r = 0; r < 8; ++r) {
        const int mm = mt * 16 + r + hf * 8;
        dst[(long)mm * HIDDEN + hh * HDIM + n] = acc[r];
    }
}

static void build_pla(PLAParams* p) {
    double xs[1001], ys[1001];
    for (int i = 0; i <= 1000; ++i) {
        xs[i] = -10.0 + (10.0 * i) / 1000.0;
        ys[i] = exp(xs[i]);
    }
    xs[1000] = 0.0; ys[1000] = 1.0;
    double itv[13];
    for (int j = 0; j <= 12; ++j) itv[j] = -10.0 + (10.0 * j) / 12.0;
    itv[12] = 0.0;
    for (int k = 0; k < 12; ++k) {
        const double a = itv[k], b = itv[k + 1];
        double sx = 0, sy = 0, sxx = 0, sxy = 0;
        int n = 0;
        for (int i = 0; i <= 1000; ++i) {
            if (xs[i] >= a && xs[i] <= b) {
                sx += xs[i]; sy += ys[i];
                sxx += xs[i] * xs[i]; sxy += xs[i] * ys[i];
                ++n;
            }
        }
        const double dn = (double)n;
        const double m = (dn * sxy - sx * sy) / (dn * sxx - sx * sx);
        const double c = (sy - m * sx) / dn;
        p->m[k] = (float)m;
        p->c[k] = (float)c;
    }
    for (int j = 0; j < 13; ++j) p->itv[j] = (float)itv[j];
}

extern "C" void kernel_launch(void* const* d_in, const int* in_sizes, int n_in,
                              void* d_out, int out_size, void* d_ws, size_t ws_size,
                              hipStream_t stream) {
    (void)in_sizes; (void)n_in; (void)out_size; (void)ws_size;

    const float* hs = (const float*)d_in[0];
    const float* Wq = (const float*)d_in[1];
    const float* bq = (const float*)d_in[2];
    const float* Wk = (const float*)d_in[3];
    const float* bk = (const float*)d_in[4];
    const float* Wv = (const float*)d_in[5];
    const float* bv = (const float*)d_in[6];
    const float* Wo = (const float*)d_in[7];
    const float* bo = (const float*)d_in[8];
    float* outp = (float*)d_out;

    const long ROWS = (long)BATCH * SEQ;
    const long BHS  = (long)BATCH * NHEAD * SEQ;

    char* w = (char*)d_ws;
    auto alloc = [&](size_t n) -> void* {
        void* p = (void*)w;
        w += (n + 255) & ~(size_t)255;
        return p;
    };

    signed char* Xq  = (signed char*)alloc(ROWS * HIDDEN);
    float*       sX  = (float*)alloc(ROWS * 4);
    signed char* Wqi = (signed char*)alloc((size_t)HIDDEN * HIDDEN);
    signed char* Wki = (signed char*)alloc((size_t)HIDDEN * HIDDEN);
    signed char* Wvi = (signed char*)alloc((size_t)HIDDEN * HIDDEN);
    signed char* Woi = (signed char*)alloc((size_t)HIDDEN * HIDDEN);
    float*       sWq = (float*)alloc(HIDDEN * 4);
    float*       sWk = (float*)alloc(HIDDEN * 4);
    float*       sWv = (float*)alloc(HIDDEN * 4);
    float*       sWo = (float*)alloc(HIDDEN * 4);
    float*       Qf  = (float*)alloc(ROWS * HIDDEN * 4);

    const size_t projBytes = (size_t)ROWS * HIDDEN * 4;
    const size_t probBytes = (size_t)BATCH * NHEAD * SEQ * SEQ * 2;
    char* big = (char*)alloc(probBytes > 2 * projBytes ? probBytes : 2 * projBytes);
    float*    Kf   = (float*)big;
    float*    Vf   = (float*)(big + projBytes);
    _Float16* ph16 = (_Float16*)big;

    signed char* qi8  = (signed char*)alloc(BHS * HDIM);
    signed char* ki8  = (signed char*)alloc(BHS * HDIM);
    float*       sq   = (float*)alloc(BHS * 4);
    float*       sk   = (float*)alloc(BHS * 4);
    _Float16*    vh16 = (_Float16*)alloc(BHS * HDIM * 2);
    signed char* ci8  = (signed char*)alloc(ROWS * HIDDEN);
    float*       sC   = (float*)alloc(ROWS * 4);
    float*       ctxf = Qf;

    PLAParams pla;
    build_pla(&pla);

    rowquant768_kernel<<<(int)ROWS, 256, 0, stream>>>(hs, Xq, sX);
    rowquant768_kernel<<<HIDDEN, 256, 0, stream>>>(Wq, Wqi, sWq);
    rowquant768_kernel<<<HIDDEN, 256, 0, stream>>>(Wk, Wki, sWk);
    rowquant768_kernel<<<HIDDEN, 256, 0, stream>>>(Wv, Wvi, sWv);
    rowquant768_kernel<<<HIDDEN, 256, 0, stream>>>(Wo, Woi, sWo);

    dim3 gProj((int)(ROWS / 16), HIDDEN / 256);
    proj_iu8_kernel<<<gProj, 128, 0, stream>>>(Xq, sX, Wqi, sWq, bq, Qf);
    proj_iu8_kernel<<<gProj, 128, 0, stream>>>(Xq, sX, Wki, sWk, bk, Kf);
    proj_iu8_kernel<<<gProj, 128, 0, stream>>>(Xq, sX, Wvi, sWv, bv, Vf);

    const int gHQ = (int)(BHS / 8);
    headquant_i8_kernel <<<gHQ, 256, 0, stream>>>(Qf, qi8, sq);
    headquant_i8_kernel <<<gHQ, 256, 0, stream>>>(Kf, ki8, sk);
    headquant_f16_kernel<<<gHQ, 256, 0, stream>>>(Vf, vh16);

    dim3 gSS(SEQ / 16, BATCH * NHEAD);
    scores_softmax_kernel<<<gSS, 32, 0, stream>>>(qi8, sq, ki8, sk, ph16, pla);

    dim3 gCtx(SEQ / 16, HDIM / 16, BATCH * NHEAD);
    ctx_f16_kernel<<<gCtx, 32, 0, stream>>>(ph16, vh16, ctxf);

    rowquant768_kernel<<<(int)ROWS, 256, 0, stream>>>(ctxf, ci8, sC);
    proj_iu8_kernel<<<gProj, 128, 0, stream>>>(ci8, sC, Woi, sWo, bo, outp);
}
